// DeepMove_81673098100758
// MI455X (gfx1250) — matrix operand order repaired and checked by reading, not compiled, not run on hardware
//
#include <hip/hip_runtime.h>
#include <math.h>

// ---------------------------------------------------------------------------
// DeepMove forward (last-step scores only), CDNA5/gfx1250.
// GEMMs: v_wmma_f32_16x16x32_bf16, one wave per 32x64 output block
// (2 M-tiles x 4 N-tiles register blocking -> 8 WMMAs per 12 b128 loads).
// ---------------------------------------------------------------------------

typedef __attribute__((ext_vector_type(16))) __bf16 v16bf;
typedef __attribute__((ext_vector_type(8)))  __bf16 v8bf;
typedef __attribute__((ext_vector_type(8)))  float  v8f;

static constexpr int B  = 128;
static constexpr int S  = 32;
static constexpr int L  = 64;
static constexpr int H  = 512;
static constexpr int G  = 3 * H;     // 1536 gate units
static constexpr int DIN = 544;      // 512 + 32
static constexpr int DLOC = 512;
static constexpr int DTIM = 32;
static constexpr int V  = 15000;
static constexpr int K2H = 2 * H;    // 1024

__device__ __forceinline__ __bf16 f2bf(float f) {
  unsigned u = __builtin_bit_cast(unsigned, f);
  unsigned r = u + 0x7FFFu + ((u >> 16) & 1u);   // round-to-nearest-even
  unsigned short h = (unsigned short)(r >> 16);
  return __builtin_bit_cast(__bf16, h);
}

__device__ __forceinline__ float sigmoidf_(float x) {
  return 1.0f / (1.0f + expf(-x));
}

// A fragment (16x32 bf16, ISA 7.12.2): lane holds row m; halves 0..7 at
// K = kk + hi*8, halves 8..15 at K = kk + 16 + hi*8.
__device__ __forceinline__ v16bf load_frag_a(const __bf16* p, int hi) {
  v8bf a0 = *(const v8bf*)(p + (hi << 3));
  v8bf a1 = *(const v8bf*)(p + 16 + (hi << 3));
  return __builtin_shufflevector(a0, a1, 0, 1, 2, 3, 4, 5, 6, 7,
                                         8, 9, 10, 11, 12, 13, 14, 15);
}

// B fragment (32x16 bf16): lane holds column n; halves 0..7 at
// K = kk + hi*8, halves 8..15 at K = kk + 16 + hi*8.
__device__ __forceinline__ v16bf load_frag_b(const __bf16* p, int hi) {
  v8bf b0 = *(const v8bf*)(p + (hi << 3));
  v8bf b1 = *(const v8bf*)(p + 16 + (hi << 3));
  return __builtin_shufflevector(b0, b1, 0, 1, 2, 3, 4, 5, 6, 7,
                                         8, 9, 10, 11, 12, 13, 14, 15);
}

// ---------------------------------------------------------------------------
// f32 -> bf16 conversion (weights)
// ---------------------------------------------------------------------------
__global__ void k_f32_to_bf16(const float* __restrict__ in,
                              __bf16* __restrict__ out, int n) {
  int i = blockIdx.x * blockDim.x + threadIdx.x;
  if (i < n) out[i] = f2bf(in[i]);
}

// ---------------------------------------------------------------------------
// Embedding gather + concat -> bf16, layout (T, B, DIN)
// idx arrays are (B, T) row-major.
// ---------------------------------------------------------------------------
__global__ void k_embed_concat(const int* __restrict__ loc,
                               const int* __restrict__ tim, int T,
                               const float* __restrict__ eloc,
                               const float* __restrict__ etim,
                               __bf16* __restrict__ out) {
  int i = blockIdx.x * blockDim.x + threadIdx.x;
  int total = T * B * DIN;
  if (i >= total) return;
  int d  = i % DIN;
  int tb = i / DIN;
  int b  = tb % B;
  int t  = tb / B;
  float v;
  if (d < DLOC) v = eloc[(size_t)loc[b * T + t] * DLOC + d];
  else          v = etim[(size_t)tim[b * T + t] * DTIM + (d - DLOC)];
  out[i] = f2bf(v);
}

// ---------------------------------------------------------------------------
// bf16 WMMA GEMM:  D[m][n] = sum_k A[m][k] * W[n][k]  (+ bias[n])
// A: (M x K) row-major bf16 (M % 32 == 0); W: (N x K) row-major bf16.
// One wave per 32x64 output block: 2 M-tiles x 4 N-tiles register blocking.
// grid = (M/32, ceil(N/64)). K % 32 == 0. Row strides 16B-aligned.
// ---------------------------------------------------------------------------
__global__ __launch_bounds__(32)
void k_gemm_bf16(const __bf16* __restrict__ A, int lda,
                 const __bf16* __restrict__ W, int ldw,
                 const float* __restrict__ bias,
                 float* __restrict__ D, int N, int K) {
  const int lane = threadIdx.x;
  const int row  = lane & 15;
  const int hi   = lane >> 4;
  const int m0   = blockIdx.x * 32;
  const int n0   = blockIdx.y * 64;

  const __bf16* Ap[2];
#pragma unroll
  for (int mi = 0; mi < 2; ++mi)
    Ap[mi] = A + (size_t)(m0 + 16 * mi + row) * lda;

  const __bf16* Wp[4];
#pragma unroll
  for (int ni = 0; ni < 4; ++ni) {
    int wrow = n0 + 16 * ni + row;
    if (wrow >= N) wrow = N - 1;         // clamp (stores guarded below)
    Wp[ni] = W + (size_t)wrow * ldw;
  }

  v8f c[2][4];
#pragma unroll
  for (int mi = 0; mi < 2; ++mi)
#pragma unroll
    for (int ni = 0; ni < 4; ++ni) c[mi][ni] = v8f{};

  for (int kk = 0; kk < K; kk += 32) {
    v16bf a[2], b[4];
#pragma unroll
    for (int mi = 0; mi < 2; ++mi) a[mi] = load_frag_a(Ap[mi] + kk, hi);
#pragma unroll
    for (int ni = 0; ni < 4; ++ni) b[ni] = load_frag_b(Wp[ni] + kk, hi);
#pragma unroll
    for (int mi = 0; mi < 2; ++mi)
#pragma unroll
      for (int ni = 0; ni < 4; ++ni)
        c[mi][ni] = __builtin_amdgcn_wmma_f32_16x16x32_bf16(
            false, a[mi], false, b[ni], (short)0, c[mi][ni], false, false);
  }

#pragma unroll
  for (int mi = 0; mi < 2; ++mi) {
#pragma unroll
    for (int ni = 0; ni < 4; ++ni) {
      int n = n0 + 16 * ni + row;
      if (n < N) {
        float bv = bias ? bias[n] : 0.0f;
#pragma unroll
        for (int j = 0; j < 8; ++j) {
          int m = m0 + 16 * mi + j + (hi << 3);  // C/D: VGPR j -> M = j + 8*hi
          D[(size_t)m * N + n] = c[mi][ni][j] + bv;
        }
      }
    }
  }
}

// ---------------------------------------------------------------------------
// Zero hidden state (f32 + bf16 copies)
// ---------------------------------------------------------------------------
__global__ void k_zero_h(float* __restrict__ h, __bf16* __restrict__ hbf, int n) {
  int i = blockIdx.x * blockDim.x + threadIdx.x;
  if (i < n) { h[i] = 0.0f; hbf[i] = f2bf(0.0f); }
}

// ---------------------------------------------------------------------------
// GRU gate update for one time step.
// xw: (B, G) = x_t @ Wih^T + b_ih   (precomputed)
// hw: (B, G) = h   @ Whh^T + b_hh   (this step's GEMM output)
// ---------------------------------------------------------------------------
__global__ void k_gru_gate(const float* __restrict__ xw,
                           const float* __restrict__ hw,
                           float* __restrict__ h, __bf16* __restrict__ hbf,
                           float* __restrict__ hist /* (B,H) slice or null */) {
  int i = blockIdx.x * blockDim.x + threadIdx.x;
  if (i >= B * H) return;
  int b = i >> 9;            // /512
  int u = i & (H - 1);
  const float* xr = xw + (size_t)b * G;
  const float* hr = hw + (size_t)b * G;
  float r = sigmoidf_(xr[u]           + hr[u]);
  float z = sigmoidf_(xr[u + H]       + hr[u + H]);
  float n = tanhf   (xr[u + 2 * H] + r * hr[u + 2 * H]);
  float hn = (1.0f - z) * n + z * h[i];
  h[i]   = hn;
  hbf[i] = f2bf(hn);
  if (hist) hist[i] = hn;
}

// ---------------------------------------------------------------------------
// Attention over encoder history for the final decoder state + concat -> bf16.
// hdec: (B,H) f32; hist: (L,B,H) f32; outcat: (B, 2H) bf16 = [hdec | context]
// One block per batch row.
// ---------------------------------------------------------------------------
__global__ __launch_bounds__(256)
void k_attn_context(const float* __restrict__ hdec,
                    const float* __restrict__ hist,
                    __bf16* __restrict__ outcat) {
  const int b   = blockIdx.x;
  const int tid = threadIdx.x;
  __shared__ float part[4][L];
  __shared__ float a[L];
  __shared__ float mx, sm;

  const float* hd = hdec + (size_t)b * H;
  const int l = tid & (L - 1);
  const int g = tid >> 6;                       // 4 groups of 128 h-elems
  const float* hl = hist + (size_t)l * B * H + (size_t)b * H;
  float p = 0.0f;
  for (int hh = g * 128; hh < g * 128 + 128; ++hh) p += hd[hh] * hl[hh];
  part[g][l] = p;
  __syncthreads();
  if (tid < L) a[tid] = part[0][tid] + part[1][tid] + part[2][tid] + part[3][tid];
  __syncthreads();
  if (tid == 0) {
    float m = a[0];
    for (int i = 1; i < L; ++i) m = fmaxf(m, a[i]);
    mx = m;
  }
  __syncthreads();
  if (tid < L) a[tid] = expf(a[tid] - mx);
  __syncthreads();
  if (tid == 0) {
    float s = 0.0f;
    for (int i = 0; i < L; ++i) s += a[i];
    sm = s;
  }
  __syncthreads();
  const float inv = 1.0f / sm;
  for (int hh = tid; hh < H; hh += 256) {
    float c = 0.0f;
    for (int ll = 0; ll < L; ++ll)
      c += a[ll] * hist[(size_t)ll * B * H + (size_t)b * H + hh];
    outcat[(size_t)b * K2H + hh]     = f2bf(hd[hh]);
    outcat[(size_t)b * K2H + H + hh] = f2bf(c * inv);
  }
}

// ---------------------------------------------------------------------------
// In-place row log-softmax over V columns; one block per row.
// ---------------------------------------------------------------------------
__global__ __launch_bounds__(256)
void k_log_softmax(float* __restrict__ y) {
  const int b = blockIdx.x;
  float* row = y + (size_t)b * V;
  __shared__ float red[256];
  float m = -INFINITY;
  for (int v = threadIdx.x; v < V; v += 256) m = fmaxf(m, row[v]);
  red[threadIdx.x] = m;
  __syncthreads();
  for (int s = 128; s > 0; s >>= 1) {
    if (threadIdx.x < s) red[threadIdx.x] = fmaxf(red[threadIdx.x], red[threadIdx.x + s]);
    __syncthreads();
  }
  m = red[0];
  __syncthreads();
  float sum = 0.0f;
  for (int v = threadIdx.x; v < V; v += 256) sum += expf(row[v] - m);
  red[threadIdx.x] = sum;
  __syncthreads();
  for (int s = 128; s > 0; s >>= 1) {
    if (threadIdx.x < s) red[threadIdx.x] += red[threadIdx.x + s];
    __syncthreads();
  }
  const float lse = m + logf(red[0]);
  for (int v = threadIdx.x; v < V; v += 256) row[v] -= lse;
}

// ---------------------------------------------------------------------------
// Host orchestration (all on `stream`; graph-capture safe).
// Workspace usage ~145 MB.
// ---------------------------------------------------------------------------
extern "C" void kernel_launch(void* const* d_in, const int* in_sizes, int n_in,
                              void* d_out, int out_size, void* d_ws, size_t ws_size,
                              hipStream_t stream) {
  const int*   cur_loc = (const int*)d_in[0];
  const int*   cur_tim = (const int*)d_in[1];
  const int*   his_loc = (const int*)d_in[2];
  const int*   his_tim = (const int*)d_in[3];
  const float* emb_loc = (const float*)d_in[4];
  const float* emb_tim = (const float*)d_in[5];
  const float* enc_Wih = (const float*)d_in[6];
  const float* enc_Whh = (const float*)d_in[7];
  const float* enc_bih = (const float*)d_in[8];
  const float* enc_bhh = (const float*)d_in[9];
  const float* dec_Wih = (const float*)d_in[10];
  const float* dec_Whh = (const float*)d_in[11];
  const float* dec_bih = (const float*)d_in[12];
  const float* dec_bhh = (const float*)d_in[13];
  const float* fc_w    = (const float*)d_in[14];
  const float* fc_b    = (const float*)d_in[15];
  float* out = (float*)d_out;                   // (B, V) log-probs

  char* ws = (char*)d_ws;
  size_t off = 0;
  auto alloc = [&](size_t bytes) -> char* {
    off = (off + 255) & ~(size_t)255;
    char* p = ws + off;
    off += bytes;
    return p;
  };

  __bf16* hx_bf     = (__bf16*)alloc((size_t)L * B * DIN * 2);
  __bf16* x_bf      = (__bf16*)alloc((size_t)S * B * DIN * 2);
  __bf16* wih_e_bf  = (__bf16*)alloc((size_t)G * DIN * 2);
  __bf16* whh_e_bf  = (__bf16*)alloc((size_t)G * H * 2);
  __bf16* wih_d_bf  = (__bf16*)alloc((size_t)G * DIN * 2);
  __bf16* whh_d_bf  = (__bf16*)alloc((size_t)G * H * 2);
  __bf16* fcw_bf    = (__bf16*)alloc((size_t)V * K2H * 2);
  float*  xw_enc    = (float*)alloc((size_t)L * B * G * 4);
  float*  xw_dec    = (float*)alloc((size_t)S * B * G * 4);
  float*  hw        = (float*)alloc((size_t)B * G * 4);
  float*  h         = (float*)alloc((size_t)B * H * 4);
  __bf16* hbf       = (__bf16*)alloc((size_t)B * H * 2);
  float*  hist      = (float*)alloc((size_t)L * B * H * 4);
  __bf16* outcat    = (__bf16*)alloc((size_t)B * K2H * 2);
  (void)ws_size; (void)in_sizes; (void)n_in; (void)out_size;

  const int T = 256;
  auto blocks = [](size_t n, int t) { return (unsigned)((n + t - 1) / t); };

  // 1) weight conversions to bf16
  k_f32_to_bf16<<<blocks((size_t)G * DIN, T), T, 0, stream>>>(enc_Wih, wih_e_bf, G * DIN);
  k_f32_to_bf16<<<blocks((size_t)G * H,   T), T, 0, stream>>>(enc_Whh, whh_e_bf, G * H);
  k_f32_to_bf16<<<blocks((size_t)G * DIN, T), T, 0, stream>>>(dec_Wih, wih_d_bf, G * DIN);
  k_f32_to_bf16<<<blocks((size_t)G * H,   T), T, 0, stream>>>(dec_Whh, whh_d_bf, G * H);
  k_f32_to_bf16<<<blocks((size_t)V * K2H, T), T, 0, stream>>>(fc_w,    fcw_bf,   V * K2H);

  // 2) embedding gathers (time-major)
  k_embed_concat<<<blocks((size_t)L * B * DIN, T), T, 0, stream>>>(
      his_loc, his_tim, L, emb_loc, emb_tim, hx_bf);
  k_embed_concat<<<blocks((size_t)S * B * DIN, T), T, 0, stream>>>(
      cur_loc, cur_tim, S, emb_loc, emb_tim, x_bf);

  // 3) batched input-side GEMMs: xw = X @ Wih^T + b_ih
  {
    dim3 ge(L * B / 32, G / 64);
    k_gemm_bf16<<<ge, 32, 0, stream>>>(hx_bf, DIN, wih_e_bf, DIN, enc_bih, xw_enc, G, DIN);
    dim3 gd(S * B / 32, G / 64);
    k_gemm_bf16<<<gd, 32, 0, stream>>>(x_bf, DIN, wih_d_bf, DIN, dec_bih, xw_dec, G, DIN);
  }

  // 4) encoder GRU scan (stores full history)
  k_zero_h<<<blocks((size_t)B * H, T), T, 0, stream>>>(h, hbf, B * H);
  {
    dim3 gh(B / 32, G / 64);
    for (int t = 0; t < L; ++t) {
      k_gemm_bf16<<<gh, 32, 0, stream>>>(hbf, H, whh_e_bf, H, enc_bhh, hw, G, H);
      k_gru_gate<<<blocks((size_t)B * H, T), T, 0, stream>>>(
          xw_enc + (size_t)t * B * G, hw, h, hbf, hist + (size_t)t * B * H);
    }
  }

  // 5) decoder GRU scan (only final hidden state needed)
  k_zero_h<<<blocks((size_t)B * H, T), T, 0, stream>>>(h, hbf, B * H);
  {
    dim3 gh(B / 32, G / 64);
    for (int t = 0; t < S; ++t) {
      k_gemm_bf16<<<gh, 32, 0, stream>>>(hbf, H, whh_d_bf, H, dec_bhh, hw, G, H);
      k_gru_gate<<<blocks((size_t)B * H, T), T, 0, stream>>>(
          xw_dec + (size_t)t * B * G, hw, h, hbf, (float*)nullptr);
    }
  }

  // 6) attention + concat for the last step only
  k_attn_context<<<B, 256, 0, stream>>>(h, hist, outcat);

  // 7) final FC (last step only): (B x 2H) @ (V x 2H)^T + fc_b -> d_out
  {
    dim3 gf(B / 32, (V + 63) / 64);
    k_gemm_bf16<<<gf, 32, 0, stream>>>(outcat, K2H, fcw_bf, K2H, fc_b, out, V, K2H);
  }

  // 8) in-place log-softmax
  k_log_softmax<<<B, 256, 0, stream>>>(out);
}
